// GPT2MHAttention_53970559041702
// MI455X (gfx1250) — matrix operand order repaired and checked by reading, not compiled, not run on hardware
//
#include <hip/hip_runtime.h>
#include <hip/hip_bf16.h>

typedef __bf16 bf16_t;
typedef __attribute__((ext_vector_type(16))) __bf16 v16bf;
typedef __attribute__((ext_vector_type(8)))  float  v8f;

namespace {
constexpr int Bb = 4, Tt = 2048, Dd = 2048, Hh = 16, Dhh = 128;
constexpr int Mtot = Bb * Tt;   // 8192
constexpr int N1 = 3 * Dd;      // 6144
constexpr float kScale = 0.08838834764831845f;  // 1/sqrt(128)
}

// ---------- WMMA fragment helpers (CDNA5 16-bit layouts, wave32) ----------

// A-matrix 16x32 bf16: lane(0..15)=row M, lane half selects K octet:
// VGPR0..3 = K {kh*8 .. kh*8+7}, VGPR4..7 = K {16+kh*8 .. 16+kh*8+7}
__device__ __forceinline__ v16bf load_fragA(const bf16_t* base, int stride, int lane) {
  const int r  = lane & 15;
  const int kh = (lane >> 4) & 1;
  const bf16_t* p = base + (size_t)r * stride + kh * 8;
  v16bf f;
#pragma unroll
  for (int i = 0; i < 8; ++i) { f[i] = p[i]; f[8 + i] = p[16 + i]; }
  return f;
}

// B-matrix 32x16 bf16: lane(0..15)=col N, lane half selects K octet:
// VGPR0..3 = K {kh*8 .. kh*8+7}, VGPR4..7 = K {16+kh*8 .. 16+kh*8+7}
__device__ __forceinline__ v16bf load_fragB(const bf16_t* base, int stride, int lane) {
  const int n  = lane & 15;
  const int kh = (lane >> 4) & 1;
  const bf16_t* p = base + (size_t)n * stride + kh * 8;
  v16bf f;
#pragma unroll
  for (int i = 0; i < 8; ++i) { f[i] = p[i]; f[8 + i] = p[16 + i]; }
  return f;
}

__device__ __forceinline__ v8f wmma_bf16(v16bf a, v16bf b, v8f c) {
  return __builtin_amdgcn_wmma_f32_16x16x32_bf16(false, a, false, b, (short)0, c,
                                                 false, false);
}

// ---------- CDNA5 async global->LDS (ASYNCcnt path, no VGPR staging) ----------
// GV mode: 64-bit global address in VGPR pair, per-lane LDS byte offset in VGPR.
__device__ __forceinline__ void async_load_b128(unsigned lds_off,
                                                const bf16_t* gptr) {
  asm volatile("global_load_async_to_lds_b128 %0, %1, off"
               :: "v"(lds_off), "v"((unsigned long long)(uintptr_t)gptr)
               : "memory");
}

__device__ __forceinline__ void wait_async0() {
  asm volatile("s_wait_asynccnt 0" ::: "memory");
}

// Raw LDS byte offset of a __shared__ object: ISA flat addressing maps the
// shared aperture so that the low 32 address bits are the LDS offset.
__device__ __forceinline__ unsigned lds_offset(const void* p) {
  return (unsigned)(uintptr_t)p;
}

// ---------- fp32 -> bf16 convert ----------
__global__ __launch_bounds__(256) void cvt_kernel(const float* __restrict__ in,
                                                  bf16_t* __restrict__ out, int n) {
  int i = blockIdx.x * blockDim.x + threadIdx.x;
  int stride = gridDim.x * blockDim.x;
  for (; i < n; i += stride) out[i] = (bf16_t)in[i];
}

// ---------- GEMM: out = A[M,K] * W[N,K]^T + bias ----------
// Double-buffered LDS tiles filled by async-to-LDS DMA; 8 waves, wave tile 64x32.
// MODE 0: QKV epilogue -> scatter bf16 into q[B,H,T,Dh], k[B,H,T,Dh], vt[B,H,Dh,T]
// MODE 1: proj epilogue -> fp32 out[M,N]
template <int MODE>
__global__ __launch_bounds__(256) void gemm_bf16_kernel(
    const bf16_t* __restrict__ A, const bf16_t* __restrict__ W,
    const float* __restrict__ bias, int K,
    float* __restrict__ outF, bf16_t* __restrict__ qO,
    bf16_t* __restrict__ kO, bf16_t* __restrict__ vtO) {
  __shared__ __align__(16) bf16_t Asm[2][128 * 40];
  __shared__ __align__(16) bf16_t Bsm[2][128 * 40];

  const int tid = threadIdx.x;
  const int wave = tid >> 5, lane = tid & 31;
  const int lr = lane & 15, lh = (lane >> 4) & 1;
  const int wm = wave & 1, wn = wave >> 1;   // 2 x 4 wave grid, wave tile 64x32
  const int bm = blockIdx.x * 128, bn = blockIdx.y * 128;

  const unsigned aB0 = lds_offset(&Asm[0][0]), aB1 = lds_offset(&Asm[1][0]);
  const unsigned bB0 = lds_offset(&Bsm[0][0]), bB1 = lds_offset(&Bsm[1][0]);

  // This thread's two (row, 16B-chunk) slots of each 128x32 tile.
  const int id0 = tid, id1 = 256 + tid;            // 0..511
  const int r0 = id0 >> 2, c0 = id0 & 3;
  const int r1 = id1 >> 2, c1 = id1 & 3;
  const unsigned l0 = (unsigned)(r0 * 40 + c0 * 8) * 2;  // LDS byte offsets
  const unsigned l1 = (unsigned)(r1 * 40 + c1 * 8) * 2;

  auto prefetch = [&](int buf, int k0) {
    const unsigned aB = buf ? aB1 : aB0;
    const unsigned bB = buf ? bB1 : bB0;
    async_load_b128(aB + l0, A + (size_t)(bm + r0) * K + k0 + c0 * 8);
    async_load_b128(aB + l1, A + (size_t)(bm + r1) * K + k0 + c1 * 8);
    async_load_b128(bB + l0, W + (size_t)(bn + r0) * K + k0 + c0 * 8);
    async_load_b128(bB + l1, W + (size_t)(bn + r1) * K + k0 + c1 * 8);
  };

  v8f acc[4][2];
#pragma unroll
  for (int i = 0; i < 4; ++i)
#pragma unroll
    for (int j = 0; j < 2; ++j) acc[i][j] = (v8f)0.0f;

  prefetch(0, 0);
  for (int k0 = 0; k0 < K; k0 += 32) {
    const int cur = (k0 >> 5) & 1;
    wait_async0();        // this wave's async writes done
    __syncthreads();      // all waves' writes done; prior reads of other buf retired
    if (k0 + 32 < K) prefetch(cur ^ 1, k0 + 32);

    const bf16_t* At = &Asm[cur][0];
    const bf16_t* Bt = &Bsm[cur][0];
    v16bf af[4], bfr[2];
#pragma unroll
    for (int i = 0; i < 4; ++i)
      af[i] = load_fragA(At + (wm * 64 + i * 16) * 40, 40, lane);
#pragma unroll
    for (int j = 0; j < 2; ++j)
      bfr[j] = load_fragB(Bt + (wn * 32 + j * 16) * 40, 40, lane);
#pragma unroll
    for (int i = 0; i < 4; ++i)
#pragma unroll
      for (int j = 0; j < 2; ++j) acc[i][j] = wmma_bf16(af[i], bfr[j], acc[i][j]);
  }

  // Epilogue. C layout: VGPR vj, lanes 0-15 -> M=vj, lanes 16-31 -> M=vj+8; N=lr.
#pragma unroll
  for (int i = 0; i < 4; ++i) {
#pragma unroll
    for (int j = 0; j < 2; ++j) {
#pragma unroll
      for (int vj = 0; vj < 8; ++vj) {
        const int mg = bm + wm * 64 + i * 16 + vj + 8 * lh;
        const int ng = bn + wn * 32 + j * 16 + lr;
        const float v = acc[i][j][vj] + bias[ng];
        if (MODE == 1) {
          outF[(size_t)mg * Dd + ng] = v;
        } else {
          const int bq = mg >> 11, t = mg & 2047;      // T = 2048
          const int seg = ng >> 11, fi = ng & 2047;    // D = 2048 per segment
          const int h = fi >> 7, dh = fi & 127;        // Dh = 128
          const bf16_t bv = (bf16_t)v;
          const size_t bhT = (size_t)(bq * Hh + h);
          if (seg == 0)      qO[(bhT * Tt + t) * Dhh + dh] = bv;
          else if (seg == 1) kO[(bhT * Tt + t) * Dhh + dh] = bv;
          else               vtO[(bhT * Dhh + dh) * Tt + t] = bv;
        }
      }
    }
  }
}

// ---------- Flash attention: one wave = 16 query rows, online softmax ----------
__global__ __launch_bounds__(256) void flash_attn_kernel(
    const bf16_t* __restrict__ Q, const bf16_t* __restrict__ Kmat,
    const bf16_t* __restrict__ Vt, bf16_t* __restrict__ Aout) {
  __shared__ __align__(16) bf16_t Plds[8 * 16 * 40];  // per-wave 16x32 P tile (+pad)

  const int tid = threadIdx.x;
  const int wave = tid >> 5, lane = tid & 31;
  const int lr = lane & 15, lh = (lane >> 4) & 1;
  const int bh = blockIdx.y;                 // b*H + h
  const int q0 = blockIdx.x * 128 + wave * 16;

  const bf16_t* qb = Q    + (size_t)bh * Tt * Dhh;
  const bf16_t* kb = Kmat + (size_t)bh * Tt * Dhh;
  const bf16_t* vb = Vt   + (size_t)bh * Dhh * Tt;
  bf16_t* pw = &Plds[wave * 16 * 40];

  // Q fragments: 4 chunks of 32 along Dh, kept resident
  v16bf qf[4];
#pragma unroll
  for (int c = 0; c < 4; ++c)
    qf[c] = load_fragA(qb + (size_t)q0 * Dhh + c * 32, Dhh, lane);

  v8f o[8];
#pragma unroll
  for (int f = 0; f < 8; ++f) o[f] = (v8f)0.0f;
  float rmax[8], rsum[8];
#pragma unroll
  for (int j = 0; j < 8; ++j) { rmax[j] = -1e30f; rsum[j] = 0.0f; }

  const int qmax = q0 + 15;
  for (int k0 = 0; k0 <= qmax; k0 += 32) {
    // S[16q x 32k] = Q x K^T  (two 16-key halves, contract Dh in 4 chunks)
    v8f s[2];
#pragma unroll
    for (int hf = 0; hf < 2; ++hf) {
      s[hf] = (v8f)0.0f;
#pragma unroll
      for (int c = 0; c < 4; ++c) {
        v16bf kf = load_fragB(kb + (size_t)(k0 + hf * 16) * Dhh + c * 32, Dhh, lane);
        s[hf] = wmma_bf16(qf[c], kf, s[hf]);
      }
    }
    // causal mask + online softmax (row reduce across 16-lane halves)
#pragma unroll
    for (int j = 0; j < 8; ++j) {
      const int m = q0 + j + 8 * lh;
      const int kg0 = k0 + lr;
      float s0 = (kg0 <= m)      ? s[0][j] * kScale : -1e30f;
      float s1 = (kg0 + 16 <= m) ? s[1][j] * kScale : -1e30f;
      float vmax = fmaxf(s0, s1);
#pragma unroll
      for (int off = 1; off < 16; off <<= 1)
        vmax = fmaxf(vmax, __shfl_xor(vmax, off, 16));
      const float nm = fmaxf(rmax[j], vmax);
      const float corr = __expf(rmax[j] - nm);
      const float p0 = __expf(s0 - nm);
      const float p1 = __expf(s1 - nm);
      float ps = p0 + p1;
#pragma unroll
      for (int off = 1; off < 16; off <<= 1) ps += __shfl_xor(ps, off, 16);
      rsum[j] = rsum[j] * corr + ps;
      rmax[j] = nm;
#pragma unroll
      for (int f = 0; f < 8; ++f) o[f][j] = o[f][j] * corr;
      pw[(j + 8 * lh) * 40 + lr] = (bf16_t)p0;          // C-layout -> LDS
      pw[(j + 8 * lh) * 40 + lr + 16] = (bf16_t)p1;
    }
    asm volatile("s_wait_dscnt 0" ::: "memory");        // wave-local LDS RAW
    v16bf pf = load_fragA(pw, 40, lane);                // reload in A-layout
    // O[16q x 128d] += P x V  (V transposed -> contiguous key runs)
#pragma unroll
    for (int f = 0; f < 8; ++f) {
      v16bf vf = load_fragB(vb + (size_t)(f * 16) * Tt + k0, Tt, lane);
      o[f] = wmma_bf16(pf, vf, o[f]);
    }
  }

  const int b = bh >> 4, h = bh & 15;
  float inv[8];
#pragma unroll
  for (int j = 0; j < 8; ++j) inv[j] = 1.0f / rsum[j];
#pragma unroll
  for (int f = 0; f < 8; ++f) {
#pragma unroll
    for (int j = 0; j < 8; ++j) {
      const int t = q0 + j + 8 * lh;
      const int dh = f * 16 + lr;
      Aout[((size_t)(b * Tt + t)) * Dd + h * Dhh + dh] = (bf16_t)(o[f][j] * inv[j]);
    }
  }
}

// ---------- host ----------
extern "C" void kernel_launch(void* const* d_in, const int* in_sizes, int n_in,
                              void* d_out, int out_size, void* d_ws, size_t ws_size,
                              hipStream_t stream) {
  (void)in_sizes; (void)n_in; (void)out_size; (void)ws_size;
  const float* x      = (const float*)d_in[0];
  const float* qkv_w  = (const float*)d_in[1];
  const float* qkv_b  = (const float*)d_in[2];
  const float* proj_w = (const float*)d_in[3];
  const float* proj_b = (const float*)d_in[4];
  float* out = (float*)d_out;

  char* ws = (char*)d_ws;
  size_t off = 0;
  auto alloc = [&](size_t elems) {
    bf16_t* p = (bf16_t*)(ws + off);
    off += elems * sizeof(bf16_t);
    return p;
  };
  bf16_t* xb    = alloc((size_t)Mtot * Dd);        // x in bf16
  bf16_t* wqkv  = alloc((size_t)N1 * Dd);          // qkv_w bf16
  bf16_t* wproj = alloc((size_t)Dd * Dd);          // proj_w bf16
  bf16_t* qT    = alloc((size_t)Bb * Hh * Tt * Dhh);   // [B,H,T,Dh]
  bf16_t* kT    = alloc((size_t)Bb * Hh * Tt * Dhh);   // [B,H,T,Dh]
  bf16_t* vtT   = alloc((size_t)Bb * Hh * Dhh * Tt);   // [B,H,Dh,T]
  bf16_t* attO  = alloc((size_t)Mtot * Dd);        // attention out bf16 [B*T, D]

  cvt_kernel<<<4096, 256, 0, stream>>>(x, xb, Mtot * Dd);
  cvt_kernel<<<4096, 256, 0, stream>>>(qkv_w, wqkv, N1 * Dd);
  cvt_kernel<<<4096, 256, 0, stream>>>(proj_w, wproj, Dd * Dd);

  dim3 g1(Mtot / 128, N1 / 128);
  gemm_bf16_kernel<0><<<g1, 256, 0, stream>>>(xb, wqkv, qkv_b, Dd,
                                              nullptr, qT, kT, vtT);

  dim3 g2(Tt / 128, Bb * Hh);
  flash_attn_kernel<<<g2, 256, 0, stream>>>(qT, kT, vtT, attO);

  dim3 g3(Mtot / 128, Dd / 128);
  gemm_bf16_kernel<1><<<g3, 256, 0, stream>>>(attO, wproj, proj_b, Dd,
                                              out, nullptr, nullptr, nullptr);
}
